// GNN_1_interaction_solubility_46024869544460
// MI455X (gfx1250) — matrix operand order repaired and checked by reading, not compiled, not run on hardware
//
#include <hip/hip_runtime.h>
#include <hip/hip_bf16.h>

// ---------------------------------------------------------------------------
// Problem constants (from the reference)
// ---------------------------------------------------------------------------
#define NU 8192
#define NV 8192
#define NE 32768
#define NB_GRAPH 256
#define DIM 128
#define NLAYER 5

typedef __bf16 bf16_t;
typedef bf16_t v16bf __attribute__((ext_vector_type(16)));
typedef float  v8f   __attribute__((ext_vector_type(8)));

union Frag16 { v16bf v; unsigned int u[8]; unsigned short h[16]; };
union FragC  { v8f v; float f[8]; };

__device__ __forceinline__ unsigned short f2bf(float x) {
  union { float f; unsigned int u; } c; c.f = x;
  unsigned int r = c.u + 0x7FFFu + ((c.u >> 16) & 1u);   // RNE
  return (unsigned short)(r >> 16);
}

__device__ __forceinline__ v8f wmma_bf16(const Frag16 &a, const Frag16 &b, v8f c) {
  return __builtin_amdgcn_wmma_f32_16x16x32_bf16(false, a.v, false, b.v,
                                                 (short)0, c, false, false);
}

// A-fragment: 16x32 bf16, row-major source [*, K].  lane<16 -> M=lane,
// K pairs {0,1..6,7,16..23}; lane>=16 -> same M set shifted, K pairs +8.
__device__ __forceinline__ void load_a(Frag16 &f, const unsigned short *A,
                                       int row_base, int K, int k0, int lane) {
  int m = lane & 15;
  int koff = (lane >> 4) * 8;
  const unsigned short *p = A + (size_t)(row_base + m) * K + k0 + koff;
#pragma unroll
  for (int v = 0; v < 8; ++v) {
    int kp = (v < 4) ? (2 * v) : (8 + 2 * v);   // 0,2,4,6,16,18,20,22
    f.u[v] = *(const unsigned int *)(p + kp);
  }
}

// B-fragment: 32x16 bf16. Source is "B transposed": BT[n][k] contiguous in k.
// lane<16 -> N=lane, K=0..7,16..23; lane>=16 -> N=lane-16, K=8..15,24..31.
__device__ __forceinline__ void load_bt(Frag16 &f, const unsigned short *BT,
                                        int col_base, int ldk, int k0, int lane) {
  int n = lane & 15;
  int kb = (lane >> 4) * 8;
  const unsigned short *p = BT + (size_t)(col_base + n) * ldk + k0 + kb;
#pragma unroll
  for (int v = 0; v < 8; ++v) f.u[v] = *(const unsigned int *)(p + ((v < 4) ? (2 * v) : (8 + 2 * v)));
}

// ---------------------------------------------------------------------------
// Generic WMMA GEMM: C[M,N] = act(A[M,K](bf16) * B + bias), B given as
// BT[N,K] bf16.  One wave per 16x16 tile, 4 waves per block.
// ---------------------------------------------------------------------------
__global__ void k_gemm_bf16(const unsigned short *A, const unsigned short *BT,
                            const float *bias, float *C, unsigned short *Cbf,
                            int M, int N, int K, int relu) {
  int lane = threadIdx.x;
  int w = blockIdx.x * blockDim.y + threadIdx.y;
  int TN = N >> 4;
  int total = (M >> 4) * TN;
  if (w >= total) return;
  int tm = w / TN, tn = w % TN;

  FragC c;
#pragma unroll
  for (int i = 0; i < 8; ++i) c.f[i] = 0.0f;

  for (int kt = 0; kt < K; kt += 32) {
    Frag16 a, b;
    load_a(a, A, tm * 16, K, kt, lane);
    load_bt(b, BT, tn * 16, K, kt, lane);
    c.v = wmma_bf16(a, b, c.v);
  }

  int n = lane & 15;
  int hi = lane >> 4;
  float bv = bias[tn * 16 + n];
#pragma unroll
  for (int v = 0; v < 8; ++v) {
    int m = tm * 16 + v + hi * 8;
    float val = c.f[v] + bv;
    if (relu) val = fmaxf(val, 0.0f);
    size_t idx = (size_t)m * N + tn * 16 + n;
    C[idx] = val;
    if (Cbf) Cbf[idx] = f2bf(val);
  }
}

// ---------------------------------------------------------------------------
// Fused interaction: for a 16-row strip of Arep, sweep Brep and accumulate
//   Prime(strip) = sum_j mask*tanh(Arep_strip . Brep_j / sqrt(D)) * Brep_j
// Abf:  [NA,128] bf16   Bbf: [NBR,128] bf16   BTbf: [128,NBR] bf16
// ---------------------------------------------------------------------------
__global__ void k_interact(const unsigned short *Abf, const unsigned short *Bbf,
                           const unsigned short *BTbf, const int *abatch,
                           const int *bbatch, float *Prime, int NA, int NBR) {
  __shared__ unsigned short sT[4][16 * 32];
  int lane = threadIdx.x;
  int w = blockIdx.x * blockDim.y + threadIdx.y;
  if (w >= (NA >> 4)) return;
  int i0 = w * 16;
  int hi = lane >> 4;
  int n = lane & 15;
  unsigned short *myT = sT[threadIdx.y];

  // strip A fragments (K = 128 -> 4 chunks), resident in registers
  Frag16 a[4];
#pragma unroll
  for (int kt = 0; kt < 4; ++kt) load_a(a[kt], Abf, i0, 128, kt * 32, lane);

  int aid[8];
#pragma unroll
  for (int v = 0; v < 8; ++v) aid[v] = abatch[i0 + v + hi * 8];

  FragC acc[8];
#pragma unroll
  for (int nt = 0; nt < 8; ++nt)
#pragma unroll
    for (int v = 0; v < 8; ++v) acc[nt].f[v] = 0.0f;

  const float scale = 0.08838834764831845f;   // 1/sqrt(128)

  for (int j = 0; j < NBR; j += 32) {
    // ---- two 16-col im tiles: T = mask * tanh(A . B^T * scale) -> LDS (bf16)
#pragma unroll
    for (int jj = 0; jj < 32; jj += 16) {
      FragC im;
#pragma unroll
      for (int z = 0; z < 8; ++z) im.f[z] = 0.0f;
#pragma unroll
      for (int kt = 0; kt < 4; ++kt) {
        Frag16 b;
        load_bt(b, Bbf, j + jj, 128, kt * 32, lane);  // Bt[k][n]=Brep[j+jj+n][k]
        im.v = wmma_bf16(a[kt], b, im.v);
      }
      int bid = bbatch[j + jj + n];
#pragma unroll
      for (int v = 0; v < 8; ++v) {
        float t = (aid[v] == bid) ? tanhf(im.f[v] * scale) : 0.0f;
        myT[(v + hi * 8) * 32 + jj + n] = f2bf(t);
      }
    }
    asm volatile("s_wait_dscnt 0" ::: "memory");   // per-wave LDS repack fence

    // ---- reload T as a 16x32 A-fragment
    Frag16 t;
    {
      int m = lane & 15;
      const unsigned short *p = myT + m * 32 + hi * 8;
#pragma unroll
      for (int v = 0; v < 8; ++v) {
        int kp = (v < 4) ? (2 * v) : (8 + 2 * v);
        t.u[v] = *(const unsigned int *)(p + kp);
      }
    }
    // ---- Prime += T(16x32) @ Brep[j:j+32, 0:128]
#pragma unroll
    for (int nt = 0; nt < 8; ++nt) {
      Frag16 b;
      load_bt(b, BTbf, nt * 16, NBR, j, lane);  // B[k][n]=BTbf[nt*16+n][j+k]
      acc[nt].v = wmma_bf16(t, b, acc[nt].v);
    }
  }

#pragma unroll
  for (int nt = 0; nt < 8; ++nt)
#pragma unroll
    for (int v = 0; v < 8; ++v)
      Prime[(size_t)(i0 + v + hi * 8) * DIM + nt * 16 + n] = acc[nt].f[v];
}

// ---------------------------------------------------------------------------
// Small helper kernels
// ---------------------------------------------------------------------------
__global__ void k_cvt_bf16(const float *X, unsigned short *Y, int n) {
  int i = blockIdx.x * 256 + threadIdx.x;
  if (i < n) Y[i] = f2bf(X[i]);
}

__global__ void k_transpose_bf16(const float *X, unsigned short *Y, int R, int C) {
  int i = blockIdx.x * 256 + threadIdx.x;
  if (i >= R * C) return;
  int r = i / C, c = i % C;
  Y[(size_t)c * R + r] = f2bf(X[i]);
}

__global__ void k_dense40_relu(const float *X, const float *W, const float *b,
                               float *Y, int N) {
  int i = blockIdx.x * 256 + threadIdx.x;
  if (i >= N * DIM) return;
  int r = i >> 7, c = i & 127;
  float acc = b[c];
#pragma unroll
  for (int k = 0; k < 40; ++k) acc += X[r * 40 + k] * W[k * DIM + c];
  Y[i] = fmaxf(acc, 0.0f);
}

__global__ void k_selfinit(const float *X, const float *ee1, const float *ee2,
                           float *AGG, int n) {
  int i = blockIdx.x * 256 + threadIdx.x;
  if (i < n) { int c = i & 127; AGG[i] = X[i] + ee1[c] + ee2[c]; }
}

__global__ void k_edges(const float *X, const int *ei, const int *ea,
                        const float *ee1, const float *ee2, float *AGG, int E_) {
  int e = blockIdx.x;
  int d = threadIdx.x;
  int s = ei[e];
  int t = ei[E_ + e];
  int a0 = ea[e * 2 + 0];
  int a1 = ea[e * 2 + 1];
  float v = X[(size_t)s * DIM + d] + ee1[a0 * DIM + d] + ee2[a1 * DIM + d];
  atomicAdd(&AGG[(size_t)t * DIM + d], v);
}

__global__ void k_bn_stats(const float *H, float *mean, float *rstd, int Nrows) {
  __shared__ float s1[256], s2[256];
  int c = blockIdx.x, t = threadIdx.x;
  float a = 0.f, b = 0.f;
  for (int r = t; r < Nrows; r += 256) {
    float v = H[(size_t)r * DIM + c];
    a += v; b += v * v;
  }
  s1[t] = a; s2[t] = b;
  __syncthreads();
  for (int off = 128; off; off >>= 1) {
    if (t < off) { s1[t] += s1[t + off]; s2[t] += s2[t + off]; }
    __syncthreads();
  }
  if (t == 0) {
    float m = s1[0] / Nrows;
    float var = s2[0] / Nrows - m * m;
    mean[c] = m;
    rstd[c] = rsqrtf(var + 1e-5f);
  }
}

__global__ void k_bn_apply(const float *H, const float *mean, const float *rstd,
                           const float *gamma, const float *beta, float *X,
                           int n, int relu) {
  int i = blockIdx.x * 256 + threadIdx.x;
  if (i >= n) return;
  int c = i & 127;
  float v = (H[i] - mean[c]) * rstd[c] * gamma[c] + beta[c];
  if (relu) v = fmaxf(v, 0.0f);
  X[i] = v;
}

__global__ void k_zero(float *p, int n) {
  int i = blockIdx.x * 256 + threadIdx.x;
  if (i < n) p[i] = 0.0f;
}

__global__ void k_pool(const float *rep, const float *prime, const int *batch,
                       float *outP, int N) {
  int i = blockIdx.x;
  int d = threadIdx.x;   // 256 threads
  int g = batch[i];
  float v = (d < DIM) ? rep[(size_t)i * DIM + d] : prime[(size_t)i * DIM + d - DIM];
  atomicAdd(&outP[(size_t)g * 2 * DIM + d], v);
}

__global__ void k_final(const float *SUP, const float *SVP, const float *W0,
                        const float *b0, const float *W1, const float *b1,
                        const float *W2, const float *b2, const float *W3,
                        const float *b3, float *out) {
  __shared__ float sU[256], sV[256], sH0[128], sH1[64], sH2[32];
  int b = blockIdx.x, t = threadIdx.x;
  sU[t] = SUP[(size_t)b * 256 + t];
  sV[t] = SVP[(size_t)b * 256 + t];
  __syncthreads();
  if (t < 128) {
    float acc = b0[t];
    for (int k = 0; k < 256; ++k) acc += sU[k] * W0[k * 128 + t];
    for (int k = 0; k < 256; ++k) acc += sV[k] * W0[(256 + k) * 128 + t];
    for (int k = 0; k < 256; ++k) acc += sU[k] * sV[k] * W0[(512 + k) * 128 + t];
    sH0[t] = fmaxf(acc, 0.0f);
  }
  __syncthreads();
  if (t < 64) {
    float acc = b1[t];
    for (int k = 0; k < 128; ++k) acc += sH0[k] * W1[k * 64 + t];
    sH1[t] = fmaxf(acc, 0.0f);
  }
  __syncthreads();
  if (t < 32) {
    float acc = b2[t];
    for (int k = 0; k < 64; ++k) acc += sH1[k] * W2[k * 32 + t];
    sH2[t] = fmaxf(acc, 0.0f);
  }
  __syncthreads();
  if (t == 0) {
    float acc = b3[0];
    for (int k = 0; k < 32; ++k) acc += sH2[k] * W3[k];
    out[b] = acc;
  }
}

// ---------------------------------------------------------------------------
// Host orchestration
// ---------------------------------------------------------------------------
extern "C" void kernel_launch(void *const *d_in, const int *in_sizes, int n_in,
                              void *d_out, int out_size, void *d_ws,
                              size_t ws_size, hipStream_t stream) {
  const float *solute_x  = (const float *)d_in[0];
  const float *solvent_x = (const float *)d_in[1];
  const int *edge_index  = (const int *)d_in[2];
  const int *edge_attr   = (const int *)d_in[3];
  const int *sbatch      = (const int *)d_in[4];
  const int *vbatch      = (const int *)d_in[5];
  // d_in[6], d_in[7]: one-hot len matrices, not needed (mask == batch equality)
  const float *emb_W   = (const float *)d_in[8];
  const float *emb_b   = (const float *)d_in[9];
  const float *gin_W1  = (const float *)d_in[10];
  const float *gin_b1  = (const float *)d_in[11];
  const float *gin_W2  = (const float *)d_in[12];
  const float *gin_b2  = (const float *)d_in[13];
  const float *ee1     = (const float *)d_in[14];   // [L,6,128]
  const float *ee2     = (const float *)d_in[15];   // [L,3,128]
  const float *bn_g    = (const float *)d_in[16];
  const float *bn_b    = (const float *)d_in[17];
  const float *solv_W1 = (const float *)d_in[18];
  const float *solv_b1 = (const float *)d_in[19];
  const float *solv_W2 = (const float *)d_in[20];
  const float *solv_b2 = (const float *)d_in[21];
  const float *out_W0  = (const float *)d_in[22];
  const float *out_b0  = (const float *)d_in[23];
  const float *out_W1  = (const float *)d_in[24];
  const float *out_b1  = (const float *)d_in[25];
  const float *out_W2  = (const float *)d_in[26];
  const float *out_b2  = (const float *)d_in[27];
  const float *out_W3  = (const float *)d_in[28];
  const float *out_b3  = (const float *)d_in[29];

  char *ws = (char *)d_ws;
  size_t off = 0;
  auto alloc = [&](size_t bytes) -> void * {
    void *p = ws + off;
    off = (off + bytes + 255) & ~(size_t)255;
    return p;
  };
  float *X     = (float *)alloc((size_t)NU * DIM * 4);
  float *AGG   = (float *)alloc((size_t)NU * DIM * 4);
  float *H1    = (float *)alloc((size_t)NU * 2 * DIM * 4);
  float *H2    = (float *)alloc((size_t)NU * DIM * 4);
  float *SOLH  = (float *)alloc((size_t)NV * DIM * 4);
  float *VREP  = (float *)alloc((size_t)NV * DIM * 4);
  float *SPRIME = (float *)alloc((size_t)NU * DIM * 4);
  float *VPRIME = (float *)alloc((size_t)NV * DIM * 4);
  unsigned short *AGGBF  = (unsigned short *)alloc((size_t)NU * DIM * 2);
  unsigned short *H1BF   = (unsigned short *)alloc((size_t)NU * 2 * DIM * 2);
  unsigned short *SOLHBF = (unsigned short *)alloc((size_t)NV * DIM * 2);
  unsigned short *SREPBF = (unsigned short *)alloc((size_t)NU * DIM * 2);
  unsigned short *VREPBF = (unsigned short *)alloc((size_t)NV * DIM * 2);
  unsigned short *SREPT  = (unsigned short *)alloc((size_t)NU * DIM * 2);
  unsigned short *VREPT  = (unsigned short *)alloc((size_t)NV * DIM * 2);
  unsigned short *W1T    = (unsigned short *)alloc((size_t)NLAYER * 256 * 128 * 2);
  unsigned short *W2T    = (unsigned short *)alloc((size_t)NLAYER * 128 * 256 * 2);
  unsigned short *SW2T   = (unsigned short *)alloc((size_t)128 * 128 * 2);
  float *MEAN = (float *)alloc(DIM * 4);
  float *RSTD = (float *)alloc(DIM * 4);
  float *SUP  = (float *)alloc((size_t)NB_GRAPH * 2 * DIM * 4);
  float *SVP  = (float *)alloc((size_t)NB_GRAPH * 2 * DIM * 4);

  dim3 wblk(32, 4);
  const int elemsNUD = NU * DIM, gNUD = (elemsNUD + 255) / 256;
  const int elems2  = 128 * 256, g2 = (elems2 + 255) / 256;

  // ---- transpose+cvt weights to bf16
  for (int l = 0; l < NLAYER; ++l) {
    k_transpose_bf16<<<g2, 256, 0, stream>>>(gin_W1 + (size_t)l * 128 * 256,
                                             W1T + (size_t)l * 256 * 128, 128, 256);
    k_transpose_bf16<<<g2, 256, 0, stream>>>(gin_W2 + (size_t)l * 256 * 128,
                                             W2T + (size_t)l * 128 * 256, 256, 128);
  }
  k_transpose_bf16<<<(128 * 128 + 255) / 256, 256, 0, stream>>>(solv_W2, SW2T, 128, 128);

  // ---- solute atom embedding
  k_dense40_relu<<<gNUD, 256, 0, stream>>>(solute_x, emb_W, emb_b, X, NU);

  // ---- GIN layers
  for (int l = 0; l < NLAYER; ++l) {
    const float *e1 = ee1 + (size_t)l * 6 * DIM;
    const float *e2 = ee2 + (size_t)l * 3 * DIM;
    k_selfinit<<<gNUD, 256, 0, stream>>>(X, e1 + 4 * DIM, e2, AGG, elemsNUD);
    k_edges<<<NE, DIM, 0, stream>>>(X, edge_index, edge_attr, e1, e2, AGG, NE);
    k_cvt_bf16<<<gNUD, 256, 0, stream>>>(AGG, AGGBF, elemsNUD);
    {
      int tiles = (NU / 16) * (2 * DIM / 16);
      k_gemm_bf16<<<(tiles + 3) / 4, wblk, 0, stream>>>(
          AGGBF, W1T + (size_t)l * 256 * 128, gin_b1 + (size_t)l * 256,
          H1, H1BF, NU, 2 * DIM, DIM, 1);
    }
    {
      int tiles = (NU / 16) * (DIM / 16);
      k_gemm_bf16<<<(tiles + 3) / 4, wblk, 0, stream>>>(
          H1BF, W2T + (size_t)l * 128 * 256, gin_b2 + (size_t)l * DIM,
          H2, nullptr, NU, DIM, 2 * DIM, 0);
    }
    k_bn_stats<<<DIM, 256, 0, stream>>>(H2, MEAN, RSTD, NU);
    k_bn_apply<<<gNUD, 256, 0, stream>>>(H2, MEAN, RSTD, bn_g + (size_t)l * DIM,
                                         bn_b + (size_t)l * DIM, X, elemsNUD,
                                         (l < NLAYER - 1) ? 1 : 0);
  }

  // ---- solvent MLP
  k_dense40_relu<<<gNUD, 256, 0, stream>>>(solvent_x, solv_W1, solv_b1, SOLH, NV);
  k_cvt_bf16<<<gNUD, 256, 0, stream>>>(SOLH, SOLHBF, NV * DIM);
  {
    int tiles = (NV / 16) * (DIM / 16);
    k_gemm_bf16<<<(tiles + 3) / 4, wblk, 0, stream>>>(SOLHBF, SW2T, solv_b2,
                                                      VREP, VREPBF, NV, DIM, DIM, 0);
  }

  // ---- bf16 + transposed copies of the reps
  k_cvt_bf16<<<gNUD, 256, 0, stream>>>(X, SREPBF, elemsNUD);
  k_transpose_bf16<<<gNUD, 256, 0, stream>>>(X, SREPT, NU, DIM);
  k_transpose_bf16<<<gNUD, 256, 0, stream>>>(VREP, VREPT, NV, DIM);

  // ---- fused interaction (both directions, same kernel, roles swapped)
  k_interact<<<NU / 64, wblk, 0, stream>>>(SREPBF, VREPBF, VREPT, sbatch, vbatch,
                                           SPRIME, NU, NV);
  k_interact<<<NV / 64, wblk, 0, stream>>>(VREPBF, SREPBF, SREPT, vbatch, sbatch,
                                           VPRIME, NV, NU);

  // ---- per-graph pooling
  int pElems = NB_GRAPH * 2 * DIM;
  k_zero<<<(pElems + 255) / 256, 256, 0, stream>>>(SUP, pElems);
  k_zero<<<(pElems + 255) / 256, 256, 0, stream>>>(SVP, pElems);
  k_pool<<<NU, 256, 0, stream>>>(X, SPRIME, sbatch, SUP, NU);
  k_pool<<<NV, 256, 0, stream>>>(VREP, VPRIME, vbatch, SVP, NV);

  // ---- output MLP
  k_final<<<NB_GRAPH, 256, 0, stream>>>(SUP, SVP, out_W0, out_b0, out_W1, out_b1,
                                        out_W2, out_b2, out_W3, out_b3,
                                        (float *)d_out);
}
